// RnnAtt_4810363373048
// MI455X (gfx1250) — matrix operand order repaired and checked by reading, not compiled, not run on hardware
//
#include <hip/hip_runtime.h>
#include <hip/hip_bf16.h>

#define T_STEPS 16
#define NN 16384
#define DD 128
#define HH 128
#define EE 262144
#define BB 8192
#define WNDW 4

typedef __bf16 bf16_t;
typedef bf16_t v16bf __attribute__((ext_vector_type(16)));
typedef float  v8f   __attribute__((ext_vector_type(8)));

// --- CDNA5 WMMA 16x16x32 bf16 operand index maps (ISA 7.12.2, wave32) ---
__device__ __forceinline__ int a_k_of(int i, int lane) {
    return (i < 8 ? i : i + 8) + ((lane & 16) ? 8 : 0);
}
__device__ __forceinline__ int b_k_of(int i, int lane) {
    return (i < 8 ? i : i + 8) + ((lane & 16) ? 8 : 0);
}

__device__ __forceinline__ v8f wmma_bf16(v16bf a, v16bf b, v8f c) {
    return __builtin_amdgcn_wmma_f32_16x16x32_bf16(false, a, false, b,
                                                   (short)0, c, false, false);
}

__device__ __forceinline__ bf16_t to_bf16(float v) { return (bf16_t)v; }
__device__ __forceinline__ bf16_t to_bf16(bf16_t v) { return v; }
__device__ __forceinline__ float to_f32(float v) { return v; }
__device__ __forceinline__ float to_f32(bf16_t v) { return (float)v; }

// ============ one-time weight pre-pack into WMMA B-fragment layout =========
// out[((nt*ksteps+ks)*32+lane)*16+i] = bf16(B(ks*32+b_k_of(i,lane), nt*16+(lane&15)))
// mode 0 (sage):  B(k,n) = k<128 ? W0[k*128+n]   : W1[(k-128)*128+n]
// mode 1 (rnn):   B(k,n) = k<256 ? W0[n*256+k]   : W1[n*128+(k-256)]
// mode 2 (proj):  B(k,n) = W0[n*128+k]
__global__ __launch_bounds__(256) void pack_b_kernel(
    const float* __restrict__ W0, const float* __restrict__ W1,
    bf16_t* __restrict__ out, int ksteps, int mode) {
    int tid = blockIdx.x * 256 + threadIdx.x;
    int total = 8 * ksteps * 32 * 16;
    if (tid >= total) return;
    int i = tid & 15;
    int lane = (tid >> 4) & 31;
    int ks = (tid >> 9) % ksteps;
    int nt = (tid >> 9) / ksteps;
    int k = ks * 32 + b_k_of(i, lane);
    int n = nt * 16 + (lane & 15);
    float v;
    if (mode == 0)      v = (k < HH) ? W0[k * HH + n] : W1[(k - HH) * HH + n];
    else if (mode == 1) v = (k < 2 * HH) ? W0[n * 2 * HH + k]
                                         : W1[n * HH + (k - 2 * HH)];
    else                v = W0[n * HH + k];
    out[tid] = (bf16_t)v;
}

// ---------------- SAGE scatter-mean (bandwidth-bound path) ----------------
// TX = float (layer-1 feats) or bf16 (layer-2 h1): bf16 halves gather reads.
template <typename TX>
__global__ __launch_bounds__(256) void scatter_kernel(
    const TX* __restrict__ X, const int* __restrict__ src,
    const int* __restrict__ dst, float* __restrict__ sum,
    float* __restrict__ cnt) {
    int tid = blockIdx.x * blockDim.x + threadIdx.x;
    int e = tid >> 5, lane = tid & 31;
    if (e >= EE) return;
    int s = src[e], d = dst[e];
    // warm GL2 for a future edge's row (gfx1250 global_prefetch path)
    int e2 = e + 32768;
    if (lane == 0 && e2 < EE)
        __builtin_prefetch(X + (size_t)src[e2] * DD, 0, 0);
    const TX* xr = X + (size_t)s * DD + lane * 4;
    float v0 = to_f32(xr[0]), v1 = to_f32(xr[1]);
    float v2 = to_f32(xr[2]), v3 = to_f32(xr[3]);
    float* p = sum + (size_t)d * DD + lane * 4;
    atomicAdd(p + 0, v0);
    atomicAdd(p + 1, v1);
    atomicAdd(p + 2, v2);
    atomicAdd(p + 3, v3);
    if (lane == 0) atomicAdd(cnt + d, 1.f);
}

// ---- SAGE layer: out[16x128 strip/wave] = relu?(X@Ws + (sum/cnt)@Wn + b) ----
// A fragment loaded once per k-step, reused across all 8 column tiles.
template <typename TX>
__global__ __launch_bounds__(256) void sage_gemm_kernel(
    const TX* __restrict__ X, const float* __restrict__ sum,
    const float* __restrict__ cnt, const bf16_t* __restrict__ Wp,
    const float* __restrict__ bias, bf16_t* __restrict__ out, int relu) {
    int wid = (blockIdx.x * blockDim.x + threadIdx.x) >> 5;  // 0..NN/16-1
    int m0 = wid * 16;
    int lane = threadIdx.x & 31;
    int am = lane & 15;
    float rinv = 1.f / fmaxf(cnt[m0 + am], 1.f);
    const TX* xrow = X + (size_t)(m0 + am) * HH;
    const float* srow = sum + (size_t)(m0 + am) * HH;
    v8f zero = {0.f, 0.f, 0.f, 0.f, 0.f, 0.f, 0.f, 0.f};
    v8f acc[8];
#pragma unroll
    for (int nt = 0; nt < 8; ++nt) acc[nt] = zero;
    for (int ks = 0; ks < 8; ++ks) {
        v16bf a;
#pragma unroll
        for (int i = 0; i < 16; ++i) {
            int k = ks * 32 + a_k_of(i, lane);
            a[i] = (k < HH) ? to_bf16(xrow[k])
                            : (bf16_t)(srow[k - HH] * rinv);
        }
#pragma unroll
        for (int nt = 0; nt < 8; ++nt) {
            v16bf b = *(const v16bf*)(Wp + (((nt * 8 + ks) * 32 + lane) << 4));
            acc[nt] = wmma_bf16(a, b, acc[nt]);
        }
    }
    int n = lane & 15;
    int mrow = (lane & 16) ? 8 : 0;
#pragma unroll
    for (int nt = 0; nt < 8; ++nt) {
#pragma unroll
        for (int j = 0; j < 8; ++j) {
            float r = acc[nt][j] + bias[nt * 16 + n];
            if (relu) r = fmaxf(r, 0.f);
            out[(size_t)(m0 + j + mrow) * HH + nt * 16 + n] = (bf16_t)r;
        }
    }
}

// ---- fused gather + row L2 normalize into states[t, b, g*H : g*H+H] ----
__global__ __launch_bounds__(128) void gather_norm_kernel(
    const bf16_t* __restrict__ h2, const int* __restrict__ idx,
    bf16_t* __restrict__ dstbase) {
    __shared__ float red[128];
    int b = blockIdx.x, tid = threadIdx.x;
    int row = idx[b];
    float v = (float)h2[(size_t)row * HH + tid];
    red[tid] = v * v;
    __syncthreads();
    for (int s = 64; s > 0; s >>= 1) {
        if (tid < s) red[tid] += red[tid + s];
        __syncthreads();
    }
    float scl = 1.f / fmaxf(sqrtf(red[0]), 1e-12f);
    dstbase[(size_t)b * (2 * HH) + tid] = (bf16_t)(v * scl);
}

// ---- RNN step: out = relu(x@Wih^T + hprev@Whh^T + bih + bhh), K=384 ----
__global__ __launch_bounds__(256) void rnn_gemm_kernel(
    const bf16_t* __restrict__ x, const bf16_t* __restrict__ hprev,
    const bf16_t* __restrict__ Wp, const float* __restrict__ bih,
    const float* __restrict__ bhh, bf16_t* __restrict__ out) {
    int wid = (blockIdx.x * blockDim.x + threadIdx.x) >> 5;  // 0..BB/16-1
    int m0 = wid * 16;
    int lane = threadIdx.x & 31;
    int am = lane & 15;
    const bf16_t* xrow = x + (size_t)(m0 + am) * 2 * HH;
    const bf16_t* hrow = hprev ? hprev + (size_t)(m0 + am) * HH : nullptr;
    v8f zero = {0.f, 0.f, 0.f, 0.f, 0.f, 0.f, 0.f, 0.f};
    v8f acc[8];
#pragma unroll
    for (int nt = 0; nt < 8; ++nt) acc[nt] = zero;
    for (int ks = 0; ks < 12; ++ks) {
        v16bf a;
#pragma unroll
        for (int i = 0; i < 16; ++i) {
            int k = ks * 32 + a_k_of(i, lane);
            bf16_t v;
            if (k < 2 * HH)   v = xrow[k];
            else              v = hrow ? hrow[k - 2 * HH] : (bf16_t)0.f;
            a[i] = v;
        }
#pragma unroll
        for (int nt = 0; nt < 8; ++nt) {
            v16bf b = *(const v16bf*)(Wp + (((nt * 12 + ks) * 32 + lane) << 4));
            acc[nt] = wmma_bf16(a, b, acc[nt]);
        }
    }
    int n = lane & 15;
    int mrow = (lane & 16) ? 8 : 0;
#pragma unroll
    for (int nt = 0; nt < 8; ++nt) {
#pragma unroll
        for (int j = 0; j < 8; ++j) {
            out[(size_t)(m0 + j + mrow) * HH + nt * 16 + n] = (bf16_t)fmaxf(
                acc[nt][j] + bih[nt * 16 + n] + bhh[nt * 16 + n], 0.f);
        }
    }
}

// ---- fold Wo/Wout/bo/bout into wf[0..127] and wf[128]=bias ----
__global__ __launch_bounds__(128) void fuse_out_kernel(
    const float* __restrict__ Wo, const float* __restrict__ bo,
    const float* __restrict__ Wout, const float* __restrict__ bout,
    float* __restrict__ wf) {
    int k = threadIdx.x;
    float acc = 0.f;
    for (int h = 0; h < HH; ++h) acc += Wout[h] * Wo[h * HH + k];
    wf[k] = acc;
    if (k == 0) {
        float bb = 0.f;
        for (int h = 0; h < HH; ++h) bb += Wout[h] * bo[h];
        wf[HH] = bb + bout[0];
    }
}

// ---- fused attention: A fragments loaded once, 96 WMMA -> LDS, softmax ----
__global__ __launch_bounds__(64) void attn_kernel(
    const bf16_t* __restrict__ rnn, const bf16_t* __restrict__ Pq,
    const bf16_t* __restrict__ Pk, const bf16_t* __restrict__ Pv,
    const float* __restrict__ bq, const float* __restrict__ bk,
    const float* __restrict__ bv, const float* __restrict__ wf,
    float* __restrict__ out) {
    __shared__ float qkv[2][3][T_STEPS][HH];
    __shared__ float vw_s[2][T_STEPS];
    const int wave = threadIdx.x >> 5;
    const int lane = threadIdx.x & 31;
    const int am = lane & 15;
    const int b = blockIdx.x * 2 + wave;
    // A = rnn[:, b, :] (16 x 128): 4 k-step fragments, loaded ONCE (raw bf16)
    v16bf afr[4];
#pragma unroll
    for (int ks = 0; ks < 4; ++ks) {
#pragma unroll
        for (int i = 0; i < 16; ++i) {
            int k = ks * 32 + a_k_of(i, lane);
            afr[ks][i] = rnn[(size_t)am * BB * HH + (size_t)b * HH + k];
        }
    }
    const bf16_t* Pp[3] = {Pq, Pk, Pv};
    const float* bp[3] = {bq, bk, bv};
    v8f zero = {0.f, 0.f, 0.f, 0.f, 0.f, 0.f, 0.f, 0.f};
#pragma unroll
    for (int p = 0; p < 3; ++p) {
        const bf16_t* P = Pp[p];
        const float* bias = bp[p];
#pragma unroll
        for (int nt = 0; nt < 8; ++nt) {
            v8f c = zero;
#pragma unroll
            for (int ks = 0; ks < 4; ++ks) {
                v16bf bfr =
                    *(const v16bf*)(P + (((nt * 4 + ks) * 32 + lane) << 4));
                c = wmma_bf16(afr[ks], bfr, c);
            }
            int n = lane & 15;
            int mrow = (lane & 16) ? 8 : 0;
#pragma unroll
            for (int j = 0; j < 8; ++j)
                qkv[wave][p][j + mrow][nt * 16 + n] = c[j] + bias[nt * 16 + n];
        }
    }
    // wave-local LDS: produced & consumed by same wave (lockstep)
    if (lane < T_STEPS) {
        float acc = 0.f;
        for (int h = 0; h < HH; ++h) acc += qkv[wave][2][lane][h] * wf[h];
        vw_s[wave][lane] = acc;  // V[s] . wfuse
    }
    if (lane < T_STEPS) {
        const int t = lane;
        const float scale = 0.08838834764831845f;  // 1/sqrt(128)
        int s0 = t - (WNDW - 1);
        if (s0 < 0) s0 = 0;
        int cw = t - s0 + 1;
        float sc[WNDW];
        float mx = -1e30f;
        for (int w = 0; w < cw; ++w) {
            int s = s0 + w;
            float d = 0.f;
            for (int h = 0; h < HH; ++h)
                d += qkv[wave][0][t][h] * qkv[wave][1][s][h];
            sc[w] = d * scale;
            mx = fmaxf(mx, sc[w]);
        }
        float Z = 0.f, y = 0.f;
        for (int w = 0; w < cw; ++w) {
            float e = __expf(sc[w] - mx);
            Z += e;
            y += e * vw_s[wave][s0 + w];
        }
        out[(size_t)t * BB + b] = y / Z + wf[HH];
    }
}

extern "C" void kernel_launch(void* const* d_in, const int* in_sizes, int n_in,
                              void* d_out, int out_size, void* d_ws,
                              size_t ws_size, hipStream_t stream) {
    (void)in_sizes; (void)n_in; (void)out_size; (void)ws_size;
    const float* feats[2] = {(const float*)d_in[0], (const float*)d_in[1]};
    const int* src[2] = {(const int*)d_in[2], (const int*)d_in[4]};
    const int* dst[2] = {(const int*)d_in[3], (const int*)d_in[5]};
    const int* idx[2] = {(const int*)d_in[6], (const int*)d_in[7]};
    const float *Ws1 = (const float*)d_in[8], *Wn1 = (const float*)d_in[9],
                *b1 = (const float*)d_in[10];
    const float *Ws2 = (const float*)d_in[11], *Wn2 = (const float*)d_in[12],
                *b2 = (const float*)d_in[13];
    const float *Wih = (const float*)d_in[14], *Whh = (const float*)d_in[15],
                *bih = (const float*)d_in[16], *bhh = (const float*)d_in[17];
    const float *Wq = (const float*)d_in[18], *Wk = (const float*)d_in[19],
                *Wv = (const float*)d_in[20];
    const float *bq = (const float*)d_in[21], *bk = (const float*)d_in[22],
                *bv = (const float*)d_in[23];
    const float *Wo = (const float*)d_in[24], *bo = (const float*)d_in[25],
                *Wout = (const float*)d_in[26], *bout = (const float*)d_in[27];

    // f32 scratch (atomic accumulators) first, then bf16 activations/weights
    float* ws = (float*)d_ws;
    float* sum = ws;    ws += (size_t)NN * HH;
    float* cntb = ws;   ws += NN;
    float* wf = ws;     ws += 136;  // 129 used, keep 32B alignment after
    bf16_t* bw = (bf16_t*)ws;
    bf16_t* h1 = bw;     bw += (size_t)NN * HH;
    bf16_t* h2 = bw;     bw += (size_t)NN * HH;
    bf16_t* states = bw; bw += (size_t)T_STEPS * BB * 2 * HH;
    bf16_t* rnn = bw;    bw += (size_t)T_STEPS * BB * HH;
    bf16_t* Psage1 = bw; bw += 8 * 8 * 32 * 16;   // K=256
    bf16_t* Psage2 = bw; bw += 8 * 8 * 32 * 16;   // K=256
    bf16_t* Prnn = bw;   bw += 8 * 12 * 32 * 16;  // K=384
    bf16_t* Pq = bw;     bw += 8 * 4 * 32 * 16;   // K=128
    bf16_t* Pk = bw;     bw += 8 * 4 * 32 * 16;
    bf16_t* Pv = bw;     bw += 8 * 4 * 32 * 16;

    dim3 b256(256);
    dim3 scat_grid((EE * 32) / 256);
    dim3 sage_grid((NN / 16) * 32 / 256);  // one wave per 16-row strip
    dim3 rnn_grid((BB / 16) * 32 / 256);

    // one-time weight packing (bf16, WMMA B-fragment layout)
    pack_b_kernel<<<(8 * 8 * 32 * 16) / 256, b256, 0, stream>>>(Ws1, Wn1,
                                                                Psage1, 8, 0);
    pack_b_kernel<<<(8 * 8 * 32 * 16) / 256, b256, 0, stream>>>(Ws2, Wn2,
                                                                Psage2, 8, 0);
    pack_b_kernel<<<(8 * 12 * 32 * 16) / 256, b256, 0, stream>>>(Wih, Whh,
                                                                 Prnn, 12, 1);
    pack_b_kernel<<<(8 * 4 * 32 * 16) / 256, b256, 0, stream>>>(Wq, nullptr,
                                                                Pq, 4, 2);
    pack_b_kernel<<<(8 * 4 * 32 * 16) / 256, b256, 0, stream>>>(Wk, nullptr,
                                                                Pk, 4, 2);
    pack_b_kernel<<<(8 * 4 * 32 * 16) / 256, b256, 0, stream>>>(Wv, nullptr,
                                                                Pv, 4, 2);
    fuse_out_kernel<<<1, 128, 0, stream>>>(Wo, bo, Wout, bout, wf);

    for (int t = 0; t < T_STEPS; ++t) {
        for (int g = 0; g < 2; ++g) {
            const float* X0 = feats[g] + (size_t)t * NN * DD;
            const int* sp = src[g] + (size_t)t * EE;
            const int* dp = dst[g] + (size_t)t * EE;
            hipMemsetAsync(sum, 0, (size_t)NN * HH * sizeof(float), stream);
            hipMemsetAsync(cntb, 0, NN * sizeof(float), stream);
            scatter_kernel<float><<<scat_grid, b256, 0, stream>>>(X0, sp, dp,
                                                                  sum, cntb);
            sage_gemm_kernel<float><<<sage_grid, b256, 0, stream>>>(
                X0, sum, cntb, Psage1, b1, h1, 1);
            hipMemsetAsync(sum, 0, (size_t)NN * HH * sizeof(float), stream);
            hipMemsetAsync(cntb, 0, NN * sizeof(float), stream);
            scatter_kernel<bf16_t><<<scat_grid, b256, 0, stream>>>(h1, sp, dp,
                                                                   sum, cntb);
            sage_gemm_kernel<bf16_t><<<sage_grid, b256, 0, stream>>>(
                h1, sum, cntb, Psage2, b2, h2, 0);
            gather_norm_kernel<<<BB, 128, 0, stream>>>(
                h2, idx[g] + (size_t)t * BB,
                states + (size_t)t * BB * 2 * HH + g * HH);
        }
    }
    for (int t = 0; t < T_STEPS; ++t) {
        rnn_gemm_kernel<<<rnn_grid, b256, 0, stream>>>(
            states + (size_t)t * BB * 2 * HH,
            t ? rnn + (size_t)(t - 1) * BB * HH : (const bf16_t*)nullptr, Prnn,
            bih, bhh, rnn + (size_t)t * BB * HH);
    }
    attn_kernel<<<BB / 2, 64, 0, stream>>>(rnn, Pq, Pk, Pv, bq, bk, bv, wf,
                                           (float*)d_out);
}
